// RNNCELL_83863531421925
// MI455X (gfx1250) — matrix operand order repaired and checked by reading, not compiled, not run on hardware
//
#include <hip/hip_runtime.h>
#include <hip/hip_bf16.h>
#include <stdint.h>

typedef __attribute__((ext_vector_type(16))) __bf16 v16bf;
typedef __attribute__((ext_vector_type(8)))  float  v8f;
typedef unsigned short u16;
typedef __attribute__((ext_vector_type(8)))  u16 u16x8;    // 16B
typedef __attribute__((ext_vector_type(16))) u16 u16x16;   // 32B

struct BfPair { u16x8 lo; u16x8 hi; };

#define B_  128
#define T_  1024
#define I_  256
#define H_  512
#define C_  10

// ---- helpers -------------------------------------------------------------

__device__ __forceinline__ u16 f2bf(float f) {           // RNE f32 -> bf16
    uint32_t u = __builtin_bit_cast(uint32_t, f);
    u += 0x7FFFu + ((u >> 16) & 1u);
    return (u16)(u >> 16);
}
__device__ __forceinline__ float bf2f(u16 h) {
    return __builtin_bit_cast(float, ((uint32_t)h) << 16);
}
__device__ __forceinline__ float sigm(float x) { return 1.0f / (1.0f + __expf(-x)); }
__device__ __forceinline__ float tanh_fast(float x) {
    float e = __expf(2.0f * x);
    return 1.0f - 2.0f / (e + 1.0f);
}

// A operand (16-bit, 16x32 MxK): lane<16 -> row=lane, K = kc*32 + {0..15}
//                                lane>=16 -> row=lane-16, K = kc*32 + {16..31}
__device__ __forceinline__ v16bf loadA_lds(const u16* buf, int width, int kc, int lane) {
    int row  = lane & 15;
    int koff = (lane >> 4) << 4;
    const u16* p = buf + row * width + kc * 32 + koff;
    BfPair pr;
    pr.lo = *(const u16x8*)(p);         // ds_load_b128
    pr.hi = *(const u16x8*)(p + 8);     // ds_load_b128
    return __builtin_bit_cast(v16bf, pr);
}

// B operand pre-swizzled in workspace: element ((tile*32 + lane)*16 + e)
__device__ __forceinline__ v16bf loadB_glb(const u16* __restrict__ wt, int tile, int lane) {
    const u16* p = wt + (((long)tile * 32 + lane) << 4);
    u16x16 v = *(const u16x16*)p;       // 32B coalesced global load
    return __builtin_bit_cast(v16bf, v);
}

__device__ __forceinline__ v8f wmma_bf16(v16bf a, v16bf b, v8f c) {
    return __builtin_amdgcn_wmma_f32_16x16x32_bf16(false, a, false, b, (short)0, c, false, false);
}

// ---- prep kernel: convert + swizzle weights, fuse biases -----------------
// W1T: [2048 x 768]  bf16  (K = 256 of W_ih1 ++ 512 of W_hh1), tiled
// W2T: [2048 x 1024] bf16  (K = 512 of W_ih2 ++ 512 of W_hh2), tiled
// Tiled layout: tile=(tn*kchunks+kc); lane l: col = tn*16+(l&15),
//               K = kc*32 + ((l>>4)<<4) + e, e in [0,16)

__global__ void lstm_prep(const float* __restrict__ Wih1, const float* __restrict__ Whh1,
                          const float* __restrict__ bih1, const float* __restrict__ bhh1,
                          const float* __restrict__ Wih2, const float* __restrict__ Whh2,
                          const float* __restrict__ bih2, const float* __restrict__ bhh2,
                          u16* __restrict__ W1T, u16* __restrict__ W2T,
                          float* __restrict__ bias1, float* __restrict__ bias2)
{
    const int N1 = 2048 * 768;
    const int N2 = 2048 * 1024;
    const int total = N1 + N2 + 2048 + 2048;
    for (int f = blockIdx.x * blockDim.x + threadIdx.x; f < total;
         f += gridDim.x * blockDim.x) {
        if (f < N1) {
            int e = f & 15, lane = (f >> 4) & 31, tile = f >> 9;
            int kc = tile % 24, tn = tile / 24;
            int col = tn * 16 + (lane & 15);
            int k   = kc * 32 + ((lane >> 4) << 4) + e;
            float v = (k < 256) ? Wih1[(long)col * 256 + k]
                                : Whh1[(long)col * 512 + (k - 256)];
            W1T[f] = f2bf(v);
        } else if (f < N1 + N2) {
            int g = f - N1;
            int e = g & 15, lane = (g >> 4) & 31, tile = g >> 9;
            int kc = tile & 31, tn = tile >> 5;
            int col = tn * 16 + (lane & 15);
            int k   = kc * 32 + ((lane >> 4) << 4) + e;
            float v = (k < 512) ? Wih2[(long)col * 512 + k]
                                : Whh2[(long)col * 512 + (k - 512)];
            W2T[g] = f2bf(v);
        } else if (f < N1 + N2 + 2048) {
            int n = f - N1 - N2;
            bias1[n] = bih1[n] + bhh1[n];
        } else {
            int n = f - N1 - N2 - 2048;
            bias2[n] = bih2[n] + bhh2[n];
        }
    }
}

// ---- persistent LSTM kernel: 8 blocks x 256 threads, 16 batch rows each ---

__global__ __launch_bounds__(256) void lstm_persistent(
    const float* __restrict__ x,
    const u16* __restrict__ W1T, const u16* __restrict__ W2T,
    const float* __restrict__ bias1, const float* __restrict__ bias2,
    const float* __restrict__ Wout, const float* __restrict__ bout,
    float* __restrict__ out)
{
    __shared__ __align__(16) u16 lx [16 * I_];   // x_t slice, bf16
    __shared__ __align__(16) u16 lh1[16 * H_];   // h1 state, bf16
    __shared__ __align__(16) u16 lh2[16 * H_];   // h2 state, bf16

    const int tid  = threadIdx.x;
    const int lane = tid & 31;
    const int wv   = tid >> 5;          // wave id 0..7 -> hidden cols [wv*64, wv*64+64)
    const int r0   = blockIdx.x * 16;   // batch rows r0..r0+15

    for (int i = tid; i < 16 * H_; i += 256) { lh1[i] = 0; lh2[i] = 0; }

    const v8f vzero = {0.f,0.f,0.f,0.f,0.f,0.f,0.f,0.f};
    v8f c1[4], c2[4], h1n[4], h2n[4];
    #pragma unroll
    for (int cc = 0; cc < 4; ++cc) { c1[cc] = vzero; c2[cc] = vzero; }

    __syncthreads();

    const float* xbase = x + (long)r0 * T_ * I_;

    for (int t = 0; t < T_; ++t) {
        // ---- stage x_t (f32 -> bf16) into LDS, coalesced; prefetch x_{t+1}
        {
            const float* xp = xbase + (long)t * I_;
            #pragma unroll
            for (int i = 0; i < (16 * I_) / 256; ++i) {
                int idx = tid + i * 256;
                int row = idx >> 8;            // I_ == 256
                int k   = idx & (I_ - 1);
                const float* p = xp + (long)row * T_ * I_ + k;
                lx[idx] = f2bf(*p);
                if (t + 1 < T_) __builtin_prefetch(p + I_, 0, 0);  // global_prefetch
            }
        }
        __syncthreads();   // lx visible; lh2 (h2_{t-1}) visible

        // ---- layer 1: gates1 = x_t@Wih1^T + h1@Whh1^T + b ----------------
        #pragma unroll
        for (int cc = 0; cc < 4; ++cc) {
            const int tn0 = wv * 4 + cc;       // N-tile within one gate block
            v8f acc[4] = {vzero, vzero, vzero, vzero};
            #pragma unroll 2
            for (int kc = 0; kc < 8; ++kc) {   // K 0..255 from x_t
                v16bf A = loadA_lds(lx, I_, kc, lane);
                #pragma unroll
                for (int q = 0; q < 4; ++q) {
                    v16bf Bq = loadB_glb(W1T, (tn0 + q * 32) * 24 + kc, lane);
                    acc[q] = wmma_bf16(A, Bq, acc[q]);
                }
            }
            #pragma unroll 4
            for (int kc = 0; kc < 16; ++kc) {  // K 256..767 from h1_{t-1}
                v16bf A = loadA_lds(lh1, H_, kc, lane);
                #pragma unroll
                for (int q = 0; q < 4; ++q) {
                    v16bf Bq = loadB_glb(W1T, (tn0 + q * 32) * 24 + 8 + kc, lane);
                    acc[q] = wmma_bf16(A, Bq, acc[q]);
                }
            }
            const int col = wv * 64 + cc * 16 + (lane & 15);
            float bi = bias1[0 * H_ + col];
            float bf = bias1[1 * H_ + col];
            float bg = bias1[2 * H_ + col];
            float bo = bias1[3 * H_ + col];
            v8f cn = c1[cc], hn;
            #pragma unroll
            for (int e = 0; e < 8; ++e) {
                float iv = sigm(acc[0][e] + bi);
                float fv = sigm(acc[1][e] + bf);
                float gv = tanh_fast(acc[2][e] + bg);
                float ov = sigm(acc[3][e] + bo);
                float cv = fv * cn[e] + iv * gv;
                cn[e] = cv;
                hn[e] = ov * tanh_fast(cv);
            }
            c1[cc] = cn;
            h1n[cc] = hn;
        }
        __syncthreads();   // all waves done reading lh1 (h1_{t-1})
        #pragma unroll
        for (int cc = 0; cc < 4; ++cc) {
            const int col = wv * 64 + cc * 16 + (lane & 15);
            const int rb  = (lane >> 4) << 3;  // C layout: elem e -> row e (+8 for hi lanes)
            #pragma unroll
            for (int e = 0; e < 8; ++e)
                lh1[(rb + e) * H_ + col] = f2bf(h1n[cc][e]);
        }
        __syncthreads();   // lh1 = h1_t visible

        // ---- layer 2: gates2 = h1_t@Wih2^T + h2_{t-1}@Whh2^T + b ---------
        #pragma unroll
        for (int cc = 0; cc < 4; ++cc) {
            const int tn0 = wv * 4 + cc;
            v8f acc[4] = {vzero, vzero, vzero, vzero};
            #pragma unroll 4
            for (int kc = 0; kc < 16; ++kc) {  // K 0..511 from h1_t
                v16bf A = loadA_lds(lh1, H_, kc, lane);
                #pragma unroll
                for (int q = 0; q < 4; ++q) {
                    v16bf Bq = loadB_glb(W2T, (tn0 + q * 32) * 32 + kc, lane);
                    acc[q] = wmma_bf16(A, Bq, acc[q]);
                }
            }
            #pragma unroll 4
            for (int kc = 0; kc < 16; ++kc) {  // K 512..1023 from h2_{t-1}
                v16bf A = loadA_lds(lh2, H_, kc, lane);
                #pragma unroll
                for (int q = 0; q < 4; ++q) {
                    v16bf Bq = loadB_glb(W2T, (tn0 + q * 32) * 32 + 16 + kc, lane);
                    acc[q] = wmma_bf16(A, Bq, acc[q]);
                }
            }
            const int col = wv * 64 + cc * 16 + (lane & 15);
            float bi = bias2[0 * H_ + col];
            float bf = bias2[1 * H_ + col];
            float bg = bias2[2 * H_ + col];
            float bo = bias2[3 * H_ + col];
            v8f cn = c2[cc], hn;
            #pragma unroll
            for (int e = 0; e < 8; ++e) {
                float iv = sigm(acc[0][e] + bi);
                float fv = sigm(acc[1][e] + bf);
                float gv = tanh_fast(acc[2][e] + bg);
                float ov = sigm(acc[3][e] + bo);
                float cv = fv * cn[e] + iv * gv;
                cn[e] = cv;
                hn[e] = ov * tanh_fast(cv);
            }
            c2[cc] = cn;
            h2n[cc] = hn;
        }
        __syncthreads();   // all waves done reading lh2 (h2_{t-1})
        #pragma unroll
        for (int cc = 0; cc < 4; ++cc) {
            const int col = wv * 64 + cc * 16 + (lane & 15);
            const int rb  = (lane >> 4) << 3;
            #pragma unroll
            for (int e = 0; e < 8; ++e)
                lh2[(rb + e) * H_ + col] = f2bf(h2n[cc][e]);
        }
        __syncthreads();   // lh2 = h2_t visible (covers next iteration)
    }

    // ---- output projection: out = h2_T @ Wout^T + bout -------------------
    for (int i = tid; i < 16 * C_; i += 256) {
        int r = i / C_, c = i % C_;
        float s = bout[c];
        const u16*  hp = lh2 + r * H_;
        const float* wp = Wout + (long)c * H_;
        for (int k = 0; k < H_; ++k) s += bf2f(hp[k]) * wp[k];
        out[(long)(r0 + r) * C_ + c] = s;
    }
}

// ---- host-side launch ----------------------------------------------------

extern "C" void kernel_launch(void* const* d_in, const int* in_sizes, int n_in,
                              void* d_out, int out_size, void* d_ws, size_t ws_size,
                              hipStream_t stream) {
    (void)in_sizes; (void)n_in; (void)out_size; (void)ws_size;

    const float* x    = (const float*)d_in[0];
    const float* Wih1 = (const float*)d_in[1];
    const float* Whh1 = (const float*)d_in[2];
    const float* bih1 = (const float*)d_in[3];
    const float* bhh1 = (const float*)d_in[4];
    const float* Wih2 = (const float*)d_in[5];
    const float* Whh2 = (const float*)d_in[6];
    const float* bih2 = (const float*)d_in[7];
    const float* bhh2 = (const float*)d_in[8];
    const float* Wout = (const float*)d_in[9];
    const float* bout = (const float*)d_in[10];

    char* ws = (char*)d_ws;
    u16*   W1T   = (u16*)(ws);                       // 2048*768  bf16 = 3,145,728 B
    u16*   W2T   = (u16*)(ws + 3145728);             // 2048*1024 bf16 = 4,194,304 B
    float* bias1 = (float*)(ws + 7340032);           // 2048 f32
    float* bias2 = (float*)(ws + 7348224);           // 2048 f32

    const int total = 2048 * 768 + 2048 * 1024 + 2048 + 2048;
    const int pblocks = (total + 255) / 256;
    lstm_prep<<<pblocks, 256, 0, stream>>>(Wih1, Whh1, bih1, bhh1,
                                           Wih2, Whh2, bih2, bhh2,
                                           W1T, W2T, bias1, bias2);

    lstm_persistent<<<B_ / 16, 256, 0, stream>>>(x, W1T, W2T, bias1, bias2,
                                                 Wout, bout, (float*)d_out);
}
